// AlignmentAwareReference_83004537962993
// MI455X (gfx1250) — matrix operand order repaired and checked by reading, not compiled, not run on hardware
//
#include <hip/hip_runtime.h>

typedef _Float16 f16;
typedef _Float16 v16h __attribute__((ext_vector_type(16)));
typedef _Float16 v8h  __attribute__((ext_vector_type(8)));
typedef _Float16 v4h  __attribute__((ext_vector_type(4)));
typedef float    v8f  __attribute__((ext_vector_type(8)));

#define BATCH 16
#define LT    1024
#define LV    512
#define LA    512
#define EDIM  256
#define TDIM  1536
#define NHEADS 4
#define HD    64

#define NEG_INF (-__builtin_inff())

// ---- fragment load helpers (match CDNA5 f16 WMMA VGPR layouts) ----
// A 16x32 f16: lanes 0-15 row M=lane hold K {0..7,16..23}; lanes 16-31 hold K {8..15,24..31}
// B 32x16 f16: lane n=lane&15 is column n; lanes 0-15 hold K {0..7,16..23}, lanes 16-31 hold K {8..15,24..31}
__device__ __forceinline__ void load8(v16h& a, int base, const float* p) {
  const float4 f0 = *(const float4*)p;
  const float4 f1 = *(const float4*)(p + 4);
  a[base+0]=(f16)f0.x; a[base+1]=(f16)f0.y; a[base+2]=(f16)f0.z; a[base+3]=(f16)f0.w;
  a[base+4]=(f16)f1.x; a[base+5]=(f16)f1.y; a[base+6]=(f16)f1.z; a[base+7]=(f16)f1.w;
}
__device__ __forceinline__ void load8(v16h& a, int base, const f16* p) {
  v8h v = *(const v8h*)p;
#pragma unroll
  for (int i = 0; i < 8; ++i) a[base+i] = v[i];
}
template <typename T>
__device__ __forceinline__ void load16(v16h& b, const T* p) { load8(b, 0, p); load8(b, 8, p + 8); }

// ---- utility kernels ----
__global__ void zero_f32(float* p, int n) {
  int i = blockIdx.x * blockDim.x + threadIdx.x;
  if (i < n) p[i] = 0.f;
}
__global__ void cvt_f32_to_f16(const float* __restrict__ src, f16* __restrict__ dst, int n) {
  int i = (blockIdx.x * blockDim.x + threadIdx.x) * 4;
  if (i < n) {
    const float4 v = *(const float4*)(src + i);
    v4h h; h[0]=(f16)v.x; h[1]=(f16)v.y; h[2]=(f16)v.z; h[3]=(f16)v.w;
    *(v4h*)(dst + i) = h;
  }
}

// ---- GEMM: Out[M,N] (f16) = X[M,K] @ W[N,K]^T + bias; f32 accumulate via WMMA ----
// W is pre-converted to f16. One wave -> 16x64 output tile (A frag reused over 4 N-subtiles).
template <typename TIn>
__global__ __launch_bounds__(256) void gemm_xwT(
    const TIn* __restrict__ X, const f16* __restrict__ W,
    const float* __restrict__ bias, f16* __restrict__ Out,
    int M, int N, int K) {
  const int wave   = (blockIdx.x * blockDim.x + threadIdx.x) >> 5;
  const int lane   = threadIdx.x & 31;
  const int nGrp   = N >> 6;
  const int tm     = wave / nGrp;
  const int ng     = wave % nGrp;
  if (tm >= (M >> 4)) return;
  const int m0     = tm << 4;
  const int n0     = ng << 6;
  const int laneLo = lane & 15;
  const int hi     = lane >> 4;

  v8f c[4] = {};
  const TIn* xrow = X + (size_t)(m0 + laneLo) * K;
  for (int k0 = 0; k0 < K; k0 += 32) {
    if (k0 + 64 < K) __builtin_prefetch(xrow + k0 + 64, 0, 1);
    v16h a;
    load8(a, 0, xrow + k0 + hi * 8);
    load8(a, 8, xrow + k0 + 16 + hi * 8);
#pragma unroll
    for (int s = 0; s < 4; ++s) {
      v16h b;
      load8(b, 0, W + (size_t)(n0 + s * 16 + laneLo) * K + k0 + hi * 8);
      load8(b, 8, W + (size_t)(n0 + s * 16 + laneLo) * K + k0 + 16 + hi * 8);
      c[s] = __builtin_amdgcn_wmma_f32_16x16x32_f16(false, a, false, b, (short)0, c[s], false, false);
    }
  }
#pragma unroll
  for (int s = 0; s < 4; ++s) {
    const int col = n0 + s * 16 + laneLo;
    const float bv = bias[col];
#pragma unroll
    for (int r = 0; r < 8; ++r) {
      const int row = m0 + hi * 8 + r;
      Out[(size_t)row * N + col] = (f16)(c[s][r] + bv);
    }
  }
}

// ---- attention reduce: out[b,q] += alpha * (1/H) * softmax(QK^T/8) @ senti ----
// one wave per (b, head, 16-query tile). Scores computed TRANSPOSED (S^T = K_tile @ Q_tile^T)
// so each lane owns 8 key-scores of one query -> softmax reduction is pure in-lane VALU.
// Lane halves keep independent online-softmax partials (keys r and 8+r), merged with a
// single shfl_xor(16) at the end. No per-tile cross-lane traffic.
__global__ __launch_bounds__(256) void attn_reduce(
    const f16* __restrict__ Q, const f16* __restrict__ Km,
    const float* __restrict__ senti, const unsigned char* __restrict__ mask,
    float* __restrict__ out, int Lq, int Lk, float alpha) {
  const int wave   = (blockIdx.x * blockDim.x + threadIdx.x) >> 5;
  const int lane   = threadIdx.x & 31;
  const int qTiles = Lq >> 4;
  const int qt = wave % qTiles;
  const int h  = (wave / qTiles) % NHEADS;
  const int b  = wave / (qTiles * NHEADS);
  if (b >= BATCH) return;
  const int laneLo = lane & 15;
  const int hi     = lane >> 4;
  const int q0     = qt << 4;

  // Q as B-fragment (column = this lane's query), loaded once per wave
  const f16* qrow = Q + ((size_t)(b * Lq + q0 + laneLo)) * EDIM + h * HD + hi * 8;
  v16h bq0, bq1;
  load8(bq0, 0, qrow); load8(bq0, 8, qrow + 16);        // d 0..31
  load8(bq1, 0, qrow + 32); load8(bq1, 8, qrow + 48);   // d 32..63

  float m = NEG_INF, l = 0.f, acc = 0.f;
  const float rscale = 0.125f;  // 1/sqrt(64)
  const size_t kbase = (size_t)b * Lk;

  for (int kt = 0; kt < (Lk >> 4); ++kt) {
    // K tile as A-fragment (row = key)
    const f16* krow = Km + (kbase + kt * 16 + laneLo) * EDIM + h * HD;
    v16h a0, a1;
    load8(a0, 0, krow + hi * 8);
    load8(a0, 8, krow + 16 + hi * 8);
    load8(a1, 0, krow + 32 + hi * 8);
    load8(a1, 8, krow + 48 + hi * 8);
    v8f c = {};
    c = __builtin_amdgcn_wmma_f32_16x16x32_f16(false, a0, false, bq0, (short)0, c, false, false);
    c = __builtin_amdgcn_wmma_f32_16x16x32_f16(false, a1, false, bq1, (short)0, c, false, false);

    // this lane's 8 keys: kt*16 + hi*8 + r   (all data in-lane)
    const size_t koff = kbase + kt * 16 + hi * 8;
    const float4 s0 = *(const float4*)(senti + koff);
    const float4 s1 = *(const float4*)(senti + koff + 4);
    const uint2  mv = *(const uint2*)(mask + koff);
    const float sen[8] = {s0.x, s0.y, s0.z, s0.w, s1.x, s1.y, s1.z, s1.w};

    float s[8];
#pragma unroll
    for (int r = 0; r < 8; ++r) {
      const unsigned mb = ((r < 4 ? mv.x : mv.y) >> ((r & 3) * 8)) & 0xffu;
      s[r] = mb ? NEG_INF : c[r] * rscale;
    }
    float vmax = s[0];
#pragma unroll
    for (int r = 1; r < 8; ++r) vmax = fmaxf(vmax, s[r]);
    const float mn  = fmaxf(m, vmax);
    const float mnf = (mn > -1e37f) ? mn : 0.f;  // guard: all-masked-so-far
    float sp = 0.f, sps = 0.f;
#pragma unroll
    for (int r = 0; r < 8; ++r) {
      const float p = __expf(s[r] - mnf);
      sp  += p;
      sps += p * sen[r];
    }
    const float corr = __expf(m - mnf);
    l   = l * corr + sp;
    acc = acc * corr + sps;
    m   = mn;
  }

  // merge the two lane-half partials (keys split 8/8) with one exchange
  const float mo = __shfl_xor(m, 16, 32);
  const float lo = __shfl_xor(l, 16, 32);
  const float ao = __shfl_xor(acc, 16, 32);
  const float mn  = fmaxf(m, mo);
  const float mnf = (mn > -1e37f) ? mn : 0.f;
  const float Lg  = l   * __expf(m - mnf) + lo * __expf(mo - mnf);
  const float Ag  = acc * __expf(m - mnf) + ao * __expf(mo - mnf);
  if (lane < 16) atomicAdd(&out[b * Lq + q0 + laneLo], alpha * 0.25f * Ag / Lg);
}

// ---- host side ----
static void gemm_f32(const float* X, const f16* W, const float* bias, f16* Out,
                     int M, int N, int K, hipStream_t s) {
  int waves = (M / 16) * (N / 64);
  gemm_xwT<float><<<dim3(waves / 8), dim3(256), 0, s>>>(X, W, bias, Out, M, N, K);
}
static void gemm_f16(const f16* X, const f16* W, const float* bias, f16* Out,
                     int M, int N, int K, hipStream_t s) {
  int waves = (M / 16) * (N / 64);
  gemm_xwT<f16><<<dim3(waves / 8), dim3(256), 0, s>>>(X, W, bias, Out, M, N, K);
}
static void cvt(const float* src, f16* dst, int n, hipStream_t s) {
  cvt_f32_to_f16<<<dim3((n / 4 + 255) / 256), dim3(256), 0, s>>>(src, dst, n);
}

extern "C" void kernel_launch(void* const* d_in, const int* in_sizes, int n_in,
                              void* d_out, int out_size, void* d_ws, size_t ws_size,
                              hipStream_t stream) {
  const float* hidden_T = (const float*)d_in[0];
  const float* hidden_V = (const float*)d_in[1];
  const float* hidden_A = (const float*)d_in[2];
  const float* senti_T  = (const float*)d_in[3];
  const float* senti_V  = (const float*)d_in[4];
  const float* senti_A  = (const float*)d_in[5];
  const unsigned char* mask_T = (const unsigned char*)d_in[6];
  const unsigned char* mask_V = (const unsigned char*)d_in[7];
  const unsigned char* mask_A = (const unsigned char*)d_in[8];
  const float* proj_t_w = (const float*)d_in[9];
  const float* proj_t_b = (const float*)d_in[10];
  const float* wq[4]; const float* wk[4]; const float* bq[4]; const float* bk[4];
  for (int i = 0; i < 4; ++i) {
    wq[i] = (const float*)d_in[11 + 4 * i + 0];
    wk[i] = (const float*)d_in[11 + 4 * i + 1];
    bq[i] = (const float*)d_in[11 + 4 * i + 2];
    bk[i] = (const float*)d_in[11 + 4 * i + 3];
  }

  // workspace layout (f16 elements)
  f16* tproj = (f16*)d_ws;                                   // 16384*256
  f16* qbuf  = tproj + (size_t)BATCH * LT * EDIM;            // 16384*256
  f16* kbuf  = qbuf  + (size_t)BATCH * LT * EDIM;            // 16384*256
  f16* pwh   = kbuf  + (size_t)BATCH * LT * EDIM;            // 256*1536
  f16* wqh[4]; f16* wkh[4];
  {
    f16* p = pwh + (size_t)EDIM * TDIM;
    for (int i = 0; i < 4; ++i) {
      wqh[i] = p; p += (size_t)EDIM * EDIM;
      wkh[i] = p; p += (size_t)EDIM * EDIM;
    }
  }

  float* outT = (float*)d_out;          // [16,1024]
  float* outV = outT + BATCH * LT;      // [16,512]
  float* outA = outV + BATCH * LV;      // [16,512]

  zero_f32<<<dim3((BATCH * (LT + LV + LA) + 255) / 256), dim3(256), 0, stream>>>(
      (float*)d_out, BATCH * (LT + LV + LA));

  // pre-convert all weight matrices to f16 (tiny, reused by every GEMM wave)
  cvt(proj_t_w, pwh, EDIM * TDIM, stream);
  for (int i = 0; i < 4; ++i) {
    cvt(wq[i], wqh[i], EDIM * EDIM, stream);
    cvt(wk[i], wkh[i], EDIM * EDIM, stream);
  }

  // T_proj = hidden_T @ proj_t_w^T + b  -> f16 [16*1024, 256]
  gemm_f32(hidden_T, pwh, proj_t_b, tproj, BATCH * LT, EDIM, TDIM, stream);

  // op 0: v2t  (Q=hidden_V, K=T_proj, keys=T, out=senti_ref_V)
  gemm_f32(hidden_V, wqh[0], bq[0], qbuf, BATCH * LV, EDIM, EDIM, stream);
  gemm_f16(tproj,    wkh[0], bk[0], kbuf, BATCH * LT, EDIM, EDIM, stream);
  attn_reduce<<<dim3(BATCH * NHEADS * (LV / 16) / 8), dim3(256), 0, stream>>>(
      qbuf, kbuf, senti_T, mask_T, outV, LV, LT, 1.0f);

  // op 1: a2t  (Q=hidden_A, K=T_proj, keys=T, out=senti_ref_A)
  gemm_f32(hidden_A, wqh[1], bq[1], qbuf, BATCH * LA, EDIM, EDIM, stream);
  gemm_f16(tproj,    wkh[1], bk[1], kbuf, BATCH * LT, EDIM, EDIM, stream);
  attn_reduce<<<dim3(BATCH * NHEADS * (LA / 16) / 8), dim3(256), 0, stream>>>(
      qbuf, kbuf, senti_T, mask_T, outA, LA, LT, 1.0f);

  // op 2: t2v  (Q=T_proj, K=hidden_V, keys=V, out += 0.5 into senti_ref_T)
  gemm_f16(tproj,    wqh[2], bq[2], qbuf, BATCH * LT, EDIM, EDIM, stream);
  gemm_f32(hidden_V, wkh[2], bk[2], kbuf, BATCH * LV, EDIM, EDIM, stream);
  attn_reduce<<<dim3(BATCH * NHEADS * (LT / 16) / 8), dim3(256), 0, stream>>>(
      qbuf, kbuf, senti_V, mask_V, outT, LT, LV, 0.5f);

  // op 3: t2a  (Q=T_proj, K=hidden_A, keys=A, out += 0.5 into senti_ref_T)
  gemm_f16(tproj,    wqh[3], bq[3], qbuf, BATCH * LT, EDIM, EDIM, stream);
  gemm_f32(hidden_A, wkh[3], bk[3], kbuf, BATCH * LA, EDIM, EDIM, stream);
  attn_reduce<<<dim3(BATCH * NHEADS * (LT / 16) / 8), dim3(256), 0, stream>>>(
      qbuf, kbuf, senti_A, mask_A, outT, LT, LA, 0.5f);
}
